// PointConvNet4_50397146251473
// MI455X (gfx1250) — matrix operand order repaired and checked by reading, not compiled, not run on hardware
//
#include <hip/hip_runtime.h>
#include <hip/hip_bf16.h>

#define BB 128
#define PP 200
#define KK 60
#define FF 4
#define HH 64
#define NN (BB*PP)
#define HP 72   // h1 LDS row stride in halves (144 bytes = multiple of 16B for b128 loads)

typedef __attribute__((ext_vector_type(16))) _Float16 v16h;
typedef __attribute__((ext_vector_type(8)))  _Float16 v8h;
typedef __attribute__((ext_vector_type(8)))  float    v8f;

// ---------------------------------------------------------------------------
// Kernel 1: per-event kNN (K=60 of 199 candidates), exact top_k ordering
// (ascending d2, ties -> lower index). One block per event, one wave per row.
// Also emits the tuple outputs pos / batch / edge_index.
// ---------------------------------------------------------------------------
__global__ __launch_bounds__(256) void knn_kernel(
    const float* __restrict__ pos,
    int* __restrict__ esrc, int* __restrict__ edst,
    float* __restrict__ pos_out, int* __restrict__ batch_out)
{
  __shared__ float sp[PP * 3];
  const int ev   = blockIdx.x;
  const int tid  = threadIdx.x;
  const int base = ev * PP;

  for (int t = tid; t < PP; t += blockDim.x) {
    float px = pos[(base + t) * 3 + 0];
    float py = pos[(base + t) * 3 + 1];
    float pz = pos[(base + t) * 3 + 2];
    sp[t * 3 + 0] = px; sp[t * 3 + 1] = py; sp[t * 3 + 2] = pz;
    pos_out[(base + t) * 3 + 0] = px;
    pos_out[(base + t) * 3 + 1] = py;
    pos_out[(base + t) * 3 + 2] = pz;
    batch_out[base + t] = ev;
  }
  __syncthreads();

  const int lane = tid & 31;
  const int wave = tid >> 5;
  const int nw   = blockDim.x >> 5;

  for (int i = wave; i < PP; i += nw) {
    const float xi = sp[i * 3 + 0];
    const float yi = sp[i * 3 + 1];
    const float zi = sp[i * 3 + 2];
    float d[7];
#pragma unroll
    for (int s = 0; s < 7; ++s) {
      const int j = lane + 32 * s;
      float dv = 1e30f;
      if (j < PP && j != i) {
        const float dx = sp[j * 3 + 0] - xi;
        const float dy = sp[j * 3 + 1] - yi;
        const float dz = sp[j * 3 + 2] - zi;
        dv = dx * dx + dy * dy + dz * dz;
      }
      d[s] = dv;
    }
    for (int k = 0; k < KK; ++k) {
      float bv = 1e30f;
      int   bj = 0x7fffffff;
#pragma unroll
      for (int s = 0; s < 7; ++s) {
        const int j = lane + 32 * s;
        if (d[s] < bv || (d[s] == bv && j < bj)) { bv = d[s]; bj = j; }
      }
#pragma unroll
      for (int off = 16; off > 0; off >>= 1) {
        const float ov = __shfl_xor(bv, off, 32);
        const int   oj = __shfl_xor(bj, off, 32);
        if (ov < bv || (ov == bv && oj < bj)) { bv = ov; bj = oj; }
      }
      if ((bj & 31) == lane) d[bj >> 5] = 1e30f;   // winner's owner retires it
      if (lane == 0) {
        esrc[(base + i) * KK + k] = base + bj;
        edst[(base + i) * KK + k] = base + i;
      }
    }
  }
}

// ---------------------------------------------------------------------------
// Kernel 2: per-node message MLP + max-aggregation, all GEMM work on
// v_wmma_f32_16x16x32_f16. One wave per node; 61 messages padded to 64
// (pad = duplicate self-message, harmless under max). 48 WMMAs per node.
// ---------------------------------------------------------------------------
__global__ __launch_bounds__(128) void pointconv_mlp_kernel(
    const float* __restrict__ x, const float* __restrict__ pos,
    const int* __restrict__ esrc,
    const float* __restrict__ W1, const float* __restrict__ b1,
    const float* __restrict__ W2, const float* __restrict__ b2,
    float* __restrict__ out)
{
  __shared__ _Float16 w1s[8 * HH];        // [k(pad 8)][n], row 7 = 0
  __shared__ _Float16 w2s[HH * HH];       // [k][n]
  __shared__ _Float16 b1s[HH];
  __shared__ _Float16 b2s[HH];
  __shared__ _Float16 h1s[4][HH * HP];    // per-wave f16 h1 tile (64 x 72)

  const int tid = threadIdx.x;
  for (int i = tid; i < 8 * HH; i += blockDim.x) {
    const int k = i >> 6, n = i & 63;
    w1s[i] = (k < 7) ? (_Float16)W1[k * HH + n] : (_Float16)0.0f;
  }
  for (int i = tid; i < HH * HH; i += blockDim.x) w2s[i] = (_Float16)W2[i];
  for (int i = tid; i < HH; i += blockDim.x) {
    b1s[i] = (_Float16)b1[i];
    b2s[i] = (_Float16)b2[i];
  }
  __syncthreads();

  const int lane = tid & 31;
  const int wib  = tid >> 5;
  _Float16* myh1 = &h1s[wib][0];
  const int nlo  = lane & 15;            // D/B column within 16-wide chunk
  const bool hiHalf = (lane & 16) != 0;

  // ---- constant B operands (ISA §7.12.2 16-bit B layout: lanes 0-15 K0..7,16..23,
  // lanes 16-31 K8..15,24..31; element e of v16h = K kb+e (+8 if e>=8)) ----
  v16h B1[4];
#pragma unroll
  for (int nc = 0; nc < 4; ++nc) {
    v16h bop = {};
    if (!hiHalf) {                         // K 0..7,16..23; only K<8 nonzero (row7=0)
      const int n = nc * 16 + lane;
#pragma unroll
      for (int e = 0; e < 8; ++e) bop[e] = w1s[e * HH + n];
    }
    B1[nc] = bop;
  }
  v16h B2[2][4];
#pragma unroll
  for (int kc = 0; kc < 2; ++kc) {
#pragma unroll
    for (int nc = 0; nc < 4; ++nc) {
      v16h bop = {};
      const int n  = nc * 16 + nlo;
      const int kb = kc * 32 + (hiHalf ? 8 : 0);
#pragma unroll
      for (int e = 0; e < 16; ++e) bop[e] = w2s[(kb + e + (e >= 8 ? 8 : 0)) * HH + n];
      B2[kc][nc] = bop;
    }
  }
  float bias1v[4], bias2v[4];
#pragma unroll
  for (int nc = 0; nc < 4; ++nc) {
    bias1v[nc] = (float)b1s[nc * 16 + nlo];
    bias2v[nc] = (float)b2s[nc * 16 + nlo];
  }

  const int waveGlobal = blockIdx.x * (blockDim.x >> 5) + wib;
  const int totalWaves = gridDim.x * (blockDim.x >> 5);

  for (int node = waveGlobal; node < NN; node += totalWaves) {
    const float pix = pos[node * 3 + 0];
    const float piy = pos[node * 3 + 1];
    const float piz = pos[node * 3 + 2];

    // ---------------- layer 1: m[64 x 7] @ W1 -> h1 (f16 in LDS) -----------
#pragma unroll
    for (int mt = 0; mt < 4; ++mt) {
      v16h A = {};                       // lanes>=16 hold K8..15,K24..31 == 0
      if (!hiHalf) {
        const int m = mt * 16 + lane;    // message row
        int j; float rx, ry, rz;
        if (m < KK) {
          j  = esrc[node * KK + m];
          rx = pos[j * 3 + 0] - pix;
          ry = pos[j * 3 + 1] - piy;
          rz = pos[j * 3 + 2] - piz;
        } else {                         // self-loop (rows 60..63 duplicate)
          j = node; rx = 0.f; ry = 0.f; rz = 0.f;
        }
        const float* xj = &x[j * FF];
        A[0] = (_Float16)xj[0]; A[1] = (_Float16)xj[1];
        A[2] = (_Float16)xj[2]; A[3] = (_Float16)xj[3];
        A[4] = (_Float16)rx;    A[5] = (_Float16)ry;   A[6] = (_Float16)rz;
      }
      const int rb = mt * 16 + (hiHalf ? 8 : 0);
#pragma unroll
      for (int nc = 0; nc < 4; ++nc) {
        v8f acc = {};
        acc = __builtin_amdgcn_wmma_f32_16x16x32_f16(
            false, A, false, B1[nc], (short)0, acc, false, false);
        const int featt = nc * 16 + nlo;
#pragma unroll
        for (int v = 0; v < 8; ++v) {
          const float hv = fmaxf(acc[v] + bias1v[nc], 0.0f);
          myh1[(rb + v) * HP + featt] = (_Float16)hv;
        }
      }
    }

    // same-wave DS store -> load ordering (HW is in-order; "memory" pins LLVM)
    asm volatile("s_wait_dscnt 0" ::: "memory");

    // ---------------- layer 2: h1[64 x 64] @ W2, max over 64 messages ------
    v8f vmax[4] = {};
#pragma unroll
    for (int mt = 0; mt < 4; ++mt) {
      const int row = mt * 16 + nlo;
      const int fo  = hiHalf ? 8 : 0;    // A layout: lo half K{0..7,16..23}
      const _Float16* rp = myh1 + row * HP;
      const v8h a00 = *(const v8h*)(rp + fo);
      const v8h a01 = *(const v8h*)(rp + fo + 16);
      const v8h a10 = *(const v8h*)(rp + fo + 32);
      const v8h a11 = *(const v8h*)(rp + fo + 48);
      const v16h A0 = __builtin_shufflevector(a00, a01,
          0,1,2,3,4,5,6,7,8,9,10,11,12,13,14,15);
      const v16h A1 = __builtin_shufflevector(a10, a11,
          0,1,2,3,4,5,6,7,8,9,10,11,12,13,14,15);
#pragma unroll
      for (int nc = 0; nc < 4; ++nc) {
        v8f acc = {};
        acc = __builtin_amdgcn_wmma_f32_16x16x32_f16(
            false, A0, false, B2[0][nc], (short)0, acc, false, false);
        acc = __builtin_amdgcn_wmma_f32_16x16x32_f16(
            false, A1, false, B2[1][nc], (short)0, acc, false, false);
#pragma unroll
        for (int v = 0; v < 8; ++v) {
          const float hv = fmaxf(acc[v] + bias2v[nc], 0.0f);
          vmax[nc][v] = fmaxf(vmax[nc][v], hv);
        }
      }
    }

    // reduce over message rows (8 VGPRs, then the two lane-halves) and store
#pragma unroll
    for (int nc = 0; nc < 4; ++nc) {
      float r = vmax[nc][0];
#pragma unroll
      for (int v = 1; v < 8; ++v) r = fmaxf(r, vmax[nc][v]);
      r = fmaxf(r, __shfl_xor(r, 16, 32));
      if (!hiHalf) out[node * HH + nc * 16 + lane] = r;
    }
  }
}

// ---------------------------------------------------------------------------
extern "C" void kernel_launch(void* const* d_in, const int* in_sizes, int n_in,
                              void* d_out, int out_size, void* d_ws, size_t ws_size,
                              hipStream_t stream) {
  (void)in_sizes; (void)n_in; (void)out_size; (void)d_ws; (void)ws_size;
  const float* x   = (const float*)d_in[0];
  const float* pos = (const float*)d_in[1];
  // d_in[2] = batch (recomputed: i / P)
  const float* W1  = (const float*)d_in[3];
  const float* b1  = (const float*)d_in[4];
  const float* W2  = (const float*)d_in[5];
  const float* b2  = (const float*)d_in[6];

  // Output tuple, concatenated flat: out[N,H] | pos[N,3] | batch[N] | edge_index[2,N*K]
  float* out       = (float*)d_out;
  float* pos_out   = out + (size_t)NN * HH;
  int*   batch_out = (int*)d_out + (size_t)NN * HH + (size_t)NN * 3;
  int*   esrc      = batch_out + NN;
  int*   edst      = esrc + (size_t)NN * KK;

  knn_kernel<<<BB, 256, 0, stream>>>(pos, esrc, edst, pos_out, batch_out);
  pointconv_mlp_kernel<<<256, 128, 0, stream>>>(x, pos, esrc, W1, b1, W2, b2, out);
}
